// MarketLayer_86732569575683
// MI455X (gfx1250) — matrix operand order repaired and checked by reading, not compiled, not run on hardware
//
#include <hip/hip_runtime.h>

#define BATCH      16384
#define NUM_AGENTS 32
#define TOPK       4
#define DIN        256
#define DOUT       256

typedef unsigned short u16;
typedef __attribute__((ext_vector_type(16))) __bf16 v16bf;
typedef __attribute__((ext_vector_type(8)))  float  v8f;

union ABPack { uint4 q[2]; v16bf v; };

#define WMMA_BF16(A, B, C) \
    __builtin_amdgcn_wmma_f32_16x16x32_bf16(false, (A).v, false, (B).v, (short)0, (C), false, false)

__device__ __forceinline__ u16 bf16_rne(float f) {
    unsigned u = __float_as_uint(f);
    unsigned r = u + 0x7FFFu + ((u >> 16) & 1u);
    return (u16)(r >> 16);
}
__device__ __forceinline__ float bf16f(u16 h) {
    return __uint_as_float(((unsigned)h) << 16);
}

// ---------------------------------------------------------------- kernel 1
__global__ void mk_init_counts(int* __restrict__ counts, int* __restrict__ cursors) {
    int t = threadIdx.x;
    if (t < NUM_AGENTS) { counts[t] = 0; cursors[t] = 0; }
}

// ---------------------------------------------------------------- kernel 2
// Split x into bf16 hi/lo planes (row-major, same shape).
__global__ __launch_bounds__(256) void mk_split_x(
        const float* __restrict__ x, u16* __restrict__ x_hi, u16* __restrict__ x_lo)
{
    size_t base = ((size_t)blockIdx.x * 256 + threadIdx.x) * 8;
    u16 hh[8], ll[8];
#pragma unroll
    for (int j = 0; j < 8; ++j) {
        float f = x[base + j];
        u16 h = bf16_rne(f);
        hh[j] = h;
        ll[j] = bf16_rne(f - bf16f(h));
    }
    uint4 uh, ul;
    uh.x = (unsigned)hh[0] | ((unsigned)hh[1] << 16);
    uh.y = (unsigned)hh[2] | ((unsigned)hh[3] << 16);
    uh.z = (unsigned)hh[4] | ((unsigned)hh[5] << 16);
    uh.w = (unsigned)hh[6] | ((unsigned)hh[7] << 16);
    ul.x = (unsigned)ll[0] | ((unsigned)ll[1] << 16);
    ul.y = (unsigned)ll[2] | ((unsigned)ll[3] << 16);
    ul.z = (unsigned)ll[4] | ((unsigned)ll[5] << 16);
    ul.w = (unsigned)ll[6] | ((unsigned)ll[7] << 16);
    *(uint4*)(x_hi + base) = uh;
    *(uint4*)(x_lo + base) = ul;
}

// ---------------------------------------------------------------- kernel 3
// Prepack W_out into exact bf16 B-operand layout:
//   dword index = [a][nt(16)][kb(8)][lane(32)][v(8)]
//   lane: N = nt*16 + (lane&15), K-off = (lane>>4)*8
//   dword v: K = kb*32 + Koff + (v>>2)*16 + 2*(v&3) (+1 in high 16 bits)
__global__ __launch_bounds__(256) void mk_pack_w(
        const float* __restrict__ W, unsigned* __restrict__ Wpk_hi,
        unsigned* __restrict__ Wpk_lo)
{
    int idx  = blockIdx.x * 256 + threadIdx.x;
    int v    = idx & 7;
    int lane = (idx >> 3) & 31;
    int kb   = (idx >> 8) & 7;
    int nt   = (idx >> 11) & 15;
    int a    = idx >> 15;
    int o  = nt * 16 + (lane & 15);
    int K0 = kb * 32 + (lane >> 4) * 8 + (v >> 2) * 16 + 2 * (v & 3);
    const float* Wa = W + ((size_t)a * DIN + K0) * DOUT + o;
    float f0 = Wa[0];
    float f1 = Wa[DOUT];
    u16 h0 = bf16_rne(f0), h1 = bf16_rne(f1);
    u16 l0 = bf16_rne(f0 - bf16f(h0)), l1 = bf16_rne(f1 - bf16f(h1));
    Wpk_hi[idx] = (unsigned)h0 | ((unsigned)h1 << 16);
    Wpk_lo[idx] = (unsigned)l0 | ((unsigned)l1 << 16);
}

// ---------------------------------------------------------------- kernel 4
// One wave per row: 32 lanes = 32 agents; bids in fp32, wave-wide top-4.
__global__ __launch_bounds__(256) void mk_bids_topk(
        const float* __restrict__ x, const float* __restrict__ w_bid,
        const float* __restrict__ b_bid,
        float* __restrict__ out_final, float* __restrict__ out_idx,
        float* __restrict__ out_bids,
        int* __restrict__ tops, int* __restrict__ counts)
{
    __shared__ float wb[NUM_AGENTS * 257];
    int tid = threadIdx.x;
    for (int i = tid; i < NUM_AGENTS * DIN; i += 256) {
        int a = i >> 8, j = i & 255;
        wb[a * 257 + j] = w_bid[i];
    }
    __syncthreads();

    int lane = tid & 31;
    int wave = tid >> 5;
    int row  = blockIdx.x * 8 + wave;

    const float4* xr   = (const float4*)(x + (size_t)row * DIN);
    const float*  wrow = &wb[lane * 257];
    float acc = 0.f;
#pragma unroll 8
    for (int j4 = 0; j4 < DIN / 4; ++j4) {
        float4 xv = xr[j4];
        int j = j4 * 4;
        acc += wrow[j] * xv.x + wrow[j + 1] * xv.y +
               wrow[j + 2] * xv.z + wrow[j + 3] * xv.w;
    }
    float bid = acc + b_bid[lane];
    out_bids[(size_t)row * NUM_AGENTS + lane] = bid;

    float4 z = make_float4(0.f, 0.f, 0.f, 0.f);
    float4* orow = (float4*)(out_final + (size_t)row * DOUT);
    orow[lane]      = z;
    orow[lane + 32] = z;

    float v = bid;
    int winners[TOPK];
#pragma unroll
    for (int k = 0; k < TOPK; ++k) {
        float bv = v; int bi = lane;
#pragma unroll
        for (int off = 16; off >= 1; off >>= 1) {
            float ov = __shfl_xor(bv, off, 32);
            int   oi = __shfl_xor(bi, off, 32);
            if (ov > bv || (ov == bv && oi < bi)) { bv = ov; bi = oi; }
        }
        winners[k] = bi;
        if (lane == bi) v = -__builtin_inff();
    }
    if (lane == 0) {
#pragma unroll
        for (int k = 0; k < TOPK; ++k) {
            int a = winners[k];
            out_idx[(size_t)row * TOPK + k] = (float)a;
            tops[(size_t)row * TOPK + k]    = a;
            atomicAdd(&counts[a], 1);
        }
    }
}

// ---------------------------------------------------------------- kernel 5
__global__ void mk_scan_offsets(const int* __restrict__ counts, int* __restrict__ offsets) {
    int a = threadIdx.x;
    if (a < NUM_AGENTS) {
        int s = 0;
        for (int j = 0; j < a; ++j) s += counts[j];
        offsets[a] = s;
    }
}

// ---------------------------------------------------------------- kernel 6
__global__ __launch_bounds__(256) void mk_fill_rows(
        const int* __restrict__ tops, const int* __restrict__ offsets,
        int* __restrict__ cursors, int* __restrict__ rowlist)
{
    int row = blockIdx.x * 256 + threadIdx.x;
#pragma unroll
    for (int k = 0; k < TOPK; ++k) {
        int a   = tops[(size_t)row * TOPK + k];
        int pos = atomicAdd(&cursors[a], 1);
        rowlist[offsets[a] + pos] = row;
    }
}

// ---------------------------------------------------------------- kernel 7
// Grouped GEMM, bf16x3 emulation of fp32. Block = (agent, 32-row tile).
// 16 waves, one 16-col N-tile each, two 16x16 accumulators per wave
// sharing the B operand. 6x v_wmma_f32_16x16x32_bf16 per 32-K block.
__global__ __launch_bounds__(512) void mk_grouped_gemm(
        const u16* __restrict__ x_hi, const u16* __restrict__ x_lo,
        const unsigned* __restrict__ Wpk_hi, const unsigned* __restrict__ Wpk_lo,
        const float* __restrict__ b_out,
        const int* __restrict__ counts, const int* __restrict__ offsets,
        const int* __restrict__ rowlist, float* __restrict__ out)
{
    __shared__ u16 xs_hi[32 * 264];
    __shared__ u16 xs_lo[32 * 264];
    __shared__ int rows_s[32];

    int agent = blockIdx.y;
    int cnt   = counts[agent];
    int base  = blockIdx.x * 32;
    if (base >= cnt) return;                  // block-uniform: EXEC stays full
    int off = offsets[agent];
    int tid = threadIdx.x;

    if (tid < 32) {
        int r = base + tid;
        rows_s[tid] = (r < cnt) ? rowlist[off + r] : -1;
    }
    __syncthreads();

    // gather 32 rows of bf16 hi/lo x into LDS (2048 uint4 chunks)
    for (int c = tid; c < 2048; c += 512) {
        int sel = c >> 10;                    // 0 = hi, 1 = lo
        int r   = (c >> 5) & 31;
        int j   = c & 31;                     // uint4 index within row
        int grow = rows_s[r];
        uint4 val = make_uint4(0u, 0u, 0u, 0u);
        const u16* src = sel ? x_lo : x_hi;
        if (grow >= 0) val = *(const uint4*)(src + (size_t)grow * DIN + j * 8);
        u16* dst = sel ? xs_lo : xs_hi;
        *(uint4*)(dst + r * 264 + j * 8) = val;
    }
    __syncthreads();

    int lane = tid & 31;
    int wave = tid >> 5;                      // N-tile id
    int m16  = lane & 15;
    int hi16 = lane >> 4;

    // B base: uint4 index = (((a*16+nt)*8 + kb)*32 + lane)*2
    size_t bbase = (((size_t)agent * 16 + wave) * 8 * 32 + lane) * 2;
    const uint4* Bh = (const uint4*)Wpk_hi + bbase;
    const uint4* Bl = (const uint4*)Wpk_lo + bbase;

    v8f acc0 = {}, acc1 = {};
#pragma unroll
    for (int kb = 0; kb < 8; ++kb) {
        int kA = kb * 32 + hi16 * 8;          // A: K chunks {kA..kA+7},{kA+16..+23}
        int o0 = m16 * 264 + kA;
        int o1 = o0 + 16 * 264;
        ABPack ah0, al0, ah1, al1, bh, bl;
        ah0.q[0] = *(const uint4*)(xs_hi + o0);
        ah0.q[1] = *(const uint4*)(xs_hi + o0 + 16);
        al0.q[0] = *(const uint4*)(xs_lo + o0);
        al0.q[1] = *(const uint4*)(xs_lo + o0 + 16);
        ah1.q[0] = *(const uint4*)(xs_hi + o1);
        ah1.q[1] = *(const uint4*)(xs_hi + o1 + 16);
        al1.q[0] = *(const uint4*)(xs_lo + o1);
        al1.q[1] = *(const uint4*)(xs_lo + o1 + 16);
        bh.q[0] = Bh[(size_t)kb * 64];
        bh.q[1] = Bh[(size_t)kb * 64 + 1];
        bl.q[0] = Bl[(size_t)kb * 64];
        bl.q[1] = Bl[(size_t)kb * 64 + 1];

        acc0 = WMMA_BF16(ah0, bh, acc0);
        acc0 = WMMA_BF16(ah0, bl, acc0);
        acc0 = WMMA_BF16(al0, bh, acc0);
        acc1 = WMMA_BF16(ah1, bh, acc1);
        acc1 = WMMA_BF16(ah1, bl, acc1);
        acc1 = WMMA_BF16(al1, bh, acc1);
    }

    int col = wave * 16 + m16;
    float bias = 0.25f * b_out[(size_t)agent * DOUT + col];
    int rb = hi16 * 8;
#pragma unroll
    for (int v = 0; v < 8; ++v) {
        int g0 = rows_s[rb + v];
        if (g0 >= 0)
            unsafeAtomicAdd(&out[(size_t)g0 * DOUT + col], 0.25f * acc0[v] + bias);
        int g1 = rows_s[16 + rb + v];
        if (g1 >= 0)
            unsafeAtomicAdd(&out[(size_t)g1 * DOUT + col], 0.25f * acc1[v] + bias);
    }
}

// ---------------------------------------------------------------- launch
extern "C" void kernel_launch(void* const* d_in, const int* in_sizes, int n_in,
                              void* d_out, int out_size, void* d_ws, size_t ws_size,
                              hipStream_t stream) {
    const float* x     = (const float*)d_in[0];
    const float* W_out = (const float*)d_in[1];
    const float* b_out = (const float*)d_in[2];
    const float* w_bid = (const float*)d_in[3];
    const float* b_bid = (const float*)d_in[4];

    float* out_final = (float*)d_out;                          // [B,256]
    float* out_idx   = out_final + (size_t)BATCH * DOUT;       // [B,4] (as float)
    float* out_bids  = out_idx + (size_t)BATCH * TOPK;         // [B,32]

    int* ws_i    = (int*)d_ws;
    int* counts  = ws_i;
    int* cursors = ws_i + 32;
    int* offsets = ws_i + 64;
    int* tops    = ws_i + 96;                                  // B*4 ints
    int* rowlist = tops + (size_t)BATCH * TOPK;                // B*4 ints
    u16* x_hi    = (u16*)(rowlist + (size_t)BATCH * TOPK);     // B*256 bf16
    u16* x_lo    = x_hi + (size_t)BATCH * DIN;                 // B*256 bf16
    unsigned* Wpk_hi = (unsigned*)(x_lo + (size_t)BATCH * DIN);    // 1M dwords
    unsigned* Wpk_lo = Wpk_hi + (size_t)NUM_AGENTS * 16 * 8 * 32 * 8;

    mk_init_counts<<<1, 32, 0, stream>>>(counts, cursors);
    mk_split_x<<<(BATCH * DIN) / (8 * 256), 256, 0, stream>>>(x, x_hi, x_lo);
    mk_pack_w<<<(NUM_AGENTS * 16 * 8 * 32 * 8) / 256, 256, 0, stream>>>(W_out, Wpk_hi, Wpk_lo);
    mk_bids_topk<<<BATCH / 8, 256, 0, stream>>>(x, w_bid, b_bid, out_final,
                                                out_idx, out_bids, tops, counts);
    mk_scan_offsets<<<1, 32, 0, stream>>>(counts, offsets);
    mk_fill_rows<<<BATCH / 256, 256, 0, stream>>>(tops, offsets, cursors, rowlist);
    dim3 grid(BATCH / 32, NUM_AGENTS);
    mk_grouped_gemm<<<grid, 512, 0, stream>>>(x_hi, x_lo, Wpk_hi, Wpk_lo, b_out,
                                              counts, offsets, rowlist, out_final);
}
